// AttentionBlock_45346264711595
// MI455X (gfx1250) — matrix operand order repaired and checked by reading, not compiled, not run on hardware
//
#include <hip/hip_runtime.h>
#include <hip/hip_bf16.h>
#include <math.h>

#define BB 8
#define SS 2048
#define DD 512
#define MM (BB * SS)

typedef __attribute__((ext_vector_type(16))) __bf16 v16bf;
typedef __attribute__((ext_vector_type(8)))  __bf16 v8bf;
typedef __attribute__((ext_vector_type(8)))  float  v8f;

union V16 { v16bf v; v8bf h[2]; };

__device__ __forceinline__ v8f wmma_bf16(v16bf a, v16bf b, v8f c) {
  // D = A(16x32 bf16) x B(32x16 bf16) + C(16x16 f32)
  return __builtin_amdgcn_wmma_f32_16x16x32_bf16(
      /*neg_a=*/false, a, /*neg_b=*/false, b,
      /*c_mod=*/(short)0, c, /*reuse_a=*/false, /*reuse_b=*/false);
}

__device__ __forceinline__ v8f v8f_zero() {
  v8f r;
#pragma unroll
  for (int i = 0; i < 8; ++i) r[i] = 0.0f;
  return r;
}

__device__ __forceinline__ float gelu_exact(float x) {
  return 0.5f * x * (1.0f + erff(x * 0.7071067811865476f));
}

// ---------------------------------------------------------------------------
// Elementwise f32 -> bf16
// ---------------------------------------------------------------------------
__global__ void f32_to_bf16_kernel(const float* __restrict__ in,
                                   __bf16* __restrict__ out, long long n) {
  long long i = (long long)blockIdx.x * blockDim.x + threadIdx.x;
  if (i < n) out[i] = (__bf16)in[i];
}

// W[K x N] f32 -> Wt[N x K] bf16 (transposed so B-fragments are contiguous)
__global__ void transpose_bf16_kernel(const float* __restrict__ in,
                                      __bf16* __restrict__ out, int K, int N) {
  long long i = (long long)blockIdx.x * blockDim.x + threadIdx.x;
  if (i < (long long)K * N) {
    int n = (int)(i / K);
    int kk = (int)(i % K);
    out[(size_t)n * K + kk] = (__bf16)in[(size_t)kk * N + n];
  }
}

// ---------------------------------------------------------------------------
// WMMA GEMM with fused epilogues.
// Block = 256 threads = 8 waves; block owns 16 rows x 512 cols.
// Wave w owns cols [64w, 64w+64): 4 C-tiles of 16x16.
// mode 0: out_bf16 = A@Wt^T + bias                         (q, k)
// mode 1: out_T_bf16[b][col][s] = A@Wt^T + bias            (v, transposed)
// mode 2: out_bf16 = LN(gelu(res + A@Wt^T + bias))         (FFN layers)
// mode 3: out_f32  = A@Wt^T + bias                          (final)
// ---------------------------------------------------------------------------
__global__ __launch_bounds__(256) void gemm_ep(
    const __bf16* __restrict__ A, const __bf16* __restrict__ Wt,
    const float* __restrict__ bias,
    const float* __restrict__ res, const float* __restrict__ gamma,
    const float* __restrict__ beta,
    __bf16* __restrict__ outb, __bf16* __restrict__ outT,
    float* __restrict__ outf,
    int M, int N, int K, int mode) {
  const int lane = threadIdx.x & 31;
  const int wv = threadIdx.x >> 5;
  const int hi = lane >> 4;
  const int ln = lane & 15;
  const int mbase = blockIdx.x * 16;
  const int nbase = wv * 64;

  v8f acc[4];
#pragma unroll
  for (int t = 0; t < 4; ++t) acc[t] = v8f_zero();

  // A fragment: row = mbase + (lane&15), K-halves split by lane-half (+0/+8, +16)
  const __bf16* arow = A + (size_t)(mbase + ln) * K + hi * 8;

  for (int kb = 0; kb < K; kb += 32) {
    V16 a;
    a.h[0] = *(const v8bf*)(arow + kb);
    a.h[1] = *(const v8bf*)(arow + kb + 16);
#pragma unroll
    for (int t = 0; t < 4; ++t) {
      // B fragment: col = nbase+t*16+ln, K-halves split by lane-half (+0/+8, +16)
      const __bf16* brow = Wt + (size_t)(nbase + t * 16 + ln) * K + kb + hi * 8;
      V16 bb;
      bb.h[0] = *(const v8bf*)(brow);
      bb.h[1] = *(const v8bf*)(brow + 16);
      acc[t] = wmma_bf16(a.v, bb.v, acc[t]);
    }
  }

  // bias (+ residual & gelu for mode 2)
#pragma unroll
  for (int t = 0; t < 4; ++t) {
#pragma unroll
    for (int e = 0; e < 8; ++e) {
      int row = mbase + e + hi * 8;
      int col = nbase + t * 16 + ln;
      float v = acc[t][e] + bias[col];
      if (mode == 2) {
        v += res[(size_t)row * N + col];
        v = gelu_exact(v);
      }
      acc[t][e] = v;
    }
  }

  if (mode == 2) {
    __shared__ float s_sum[16][8];
    __shared__ float s_sq[16][8];
    float ps[8], pq[8];
#pragma unroll
    for (int e = 0; e < 8; ++e) {
      float s = 0.0f, q = 0.0f;
#pragma unroll
      for (int t = 0; t < 4; ++t) { float v = acc[t][e]; s += v; q += v * v; }
#pragma unroll
      for (int m = 1; m < 16; m <<= 1) {
        s += __shfl_xor(s, m, 32);
        q += __shfl_xor(q, m, 32);
      }
      ps[e] = s; pq[e] = q;
    }
    if (ln == 0) {
#pragma unroll
      for (int e = 0; e < 8; ++e) {
        s_sum[e + hi * 8][wv] = ps[e];
        s_sq[e + hi * 8][wv] = pq[e];
      }
    }
    __syncthreads();
#pragma unroll
    for (int e = 0; e < 8; ++e) {
      int r = e + hi * 8;
      float s = 0.0f, q = 0.0f;
#pragma unroll
      for (int w = 0; w < 8; ++w) { s += s_sum[r][w]; q += s_sq[r][w]; }
      float mean = s / (float)N;
      float var = q / (float)N - mean * mean;
      float rstd = rsqrtf(var + 1e-5f);
#pragma unroll
      for (int t = 0; t < 4; ++t) {
        int col = nbase + t * 16 + ln;
        float v = (acc[t][e] - mean) * rstd * gamma[col] + beta[col];
        outb[(size_t)(mbase + r) * N + col] = (__bf16)v;
      }
    }
    return;
  }

#pragma unroll
  for (int t = 0; t < 4; ++t) {
#pragma unroll
    for (int e = 0; e < 8; ++e) {
      int row = mbase + e + hi * 8;
      int col = nbase + t * 16 + ln;
      float v = acc[t][e];
      if (mode == 0) {
        outb[(size_t)row * N + col] = (__bf16)v;
      } else if (mode == 1) {
        int bidx = row / SS;
        int s = row - bidx * SS;
        outT[((size_t)bidx * DD + col) * SS + s] = (__bf16)v;
      } else {  // mode 3
        outf[(size_t)row * N + col] = v;
      }
    }
  }
}

// ---------------------------------------------------------------------------
// Flash attention. Block = 64 threads (2 waves) per 16-query-row tile.
// Per 64-key iteration: wave w computes scores for keys [32w,32w+32) over the
// full D=512 reduction; softmax stats + exp(P) exchanged via LDS; each wave
// then accumulates P@V into its 256-column half of the output (128 acc VGPRs).
// ---------------------------------------------------------------------------
__global__ __launch_bounds__(64) void attn_kernel(
    const __bf16* __restrict__ q, const __bf16* __restrict__ k,
    const __bf16* __restrict__ vT, float* __restrict__ att) {
  const int lane = threadIdx.x & 31;
  const int wv = threadIdx.x >> 5;  // 0..1
  const int hi = lane >> 4;
  const int ln = lane & 15;
  const int bidx = blockIdx.x >> 7;   // batch
  const int qt = blockIdx.x & 127;    // query tile
  const int qbase = qt * 16;
  const float scale = 0.04419417382415922f;  // 1/sqrt(512)

  __shared__ __align__(16) __bf16 s_p[16][64];
  __shared__ float s_max[2][16];
  __shared__ float s_sum[2][16];

  v8f acc[16];
#pragma unroll
  for (int t = 0; t < 16; ++t) acc[t] = v8f_zero();
  float m_run[8], l_run[8];
#pragma unroll
  for (int e = 0; e < 8; ++e) { m_run[e] = -INFINITY; l_run[e] = 0.0f; }

  const __bf16* qrow = q + ((size_t)bidx * SS + qbase + ln) * DD + hi * 8;

  for (int kb0 = 0; kb0 < SS; kb0 += 64) {
    const int keyw = kb0 + wv * 32;
    // --- scores: two 16x16 tiles (32 keys), full-D K-reduction ---
    v8f sc[2];
    sc[0] = v8f_zero(); sc[1] = v8f_zero();
    for (int dk = 0; dk < DD; dk += 32) {
      V16 a;
      a.h[0] = *(const v8bf*)(qrow + dk);
      a.h[1] = *(const v8bf*)(qrow + dk + 16);
#pragma unroll
      for (int t = 0; t < 2; ++t) {
        const __bf16* krow =
            k + ((size_t)bidx * SS + keyw + t * 16 + ln) * DD + dk + hi * 8;
        V16 bb;
        bb.h[0] = *(const v8bf*)(krow);
        bb.h[1] = *(const v8bf*)(krow + 16);
        sc[t] = wmma_bf16(a.v, bb.v, sc[t]);
      }
    }
#pragma unroll
    for (int t = 0; t < 2; ++t)
#pragma unroll
      for (int e = 0; e < 8; ++e) sc[t][e] *= scale;

    // --- local row max (reduce over 16-lane group = 16 key-columns) ---
    float mx[8];
#pragma unroll
    for (int e = 0; e < 8; ++e) {
      float m = fmaxf(sc[0][e], sc[1][e]);
#pragma unroll
      for (int msk = 1; msk < 16; msk <<= 1) m = fmaxf(m, __shfl_xor(m, msk, 32));
      mx[e] = m;
    }
    if (ln == 0) {
#pragma unroll
      for (int e = 0; e < 8; ++e) s_max[wv][e + hi * 8] = mx[e];
    }
    __syncthreads();
    float m_new[8], alpha[8];
#pragma unroll
    for (int e = 0; e < 8; ++e) {
      float om = fmaxf(s_max[0][e + hi * 8], s_max[1][e + hi * 8]);
      m_new[e] = fmaxf(m_run[e], om);
      alpha[e] = __expf(m_run[e] - m_new[e]);
      m_run[e] = m_new[e];
    }
    // --- p = exp(s - m_new), row sums, publish P ---
    float rs[8];
#pragma unroll
    for (int e = 0; e < 8; ++e) {
      float p0 = __expf(sc[0][e] - m_new[e]);
      float p1 = __expf(sc[1][e] - m_new[e]);
      sc[0][e] = p0; sc[1][e] = p1;
      float s = p0 + p1;
#pragma unroll
      for (int msk = 1; msk < 16; msk <<= 1) s += __shfl_xor(s, msk, 32);
      rs[e] = s;
    }
    if (ln == 0) {
#pragma unroll
      for (int e = 0; e < 8; ++e) s_sum[wv][e + hi * 8] = rs[e];
    }
#pragma unroll
    for (int t = 0; t < 2; ++t)
#pragma unroll
      for (int e = 0; e < 8; ++e)
        s_p[e + hi * 8][wv * 32 + t * 16 + ln] = (__bf16)sc[t][e];
    __syncthreads();
#pragma unroll
    for (int e = 0; e < 8; ++e)
      l_run[e] = l_run[e] * alpha[e] + s_sum[0][e + hi * 8] + s_sum[1][e + hi * 8];
    // --- rescale running accumulator ---
#pragma unroll
    for (int t = 0; t < 16; ++t)
#pragma unroll
      for (int e = 0; e < 8; ++e) acc[t][e] *= alpha[e];
    // --- P @ V over this iteration's 64 keys ---
    V16 a0, a1;
    a0.h[0] = *(const v8bf*)&s_p[ln][hi * 8];
    a0.h[1] = *(const v8bf*)&s_p[ln][hi * 8 + 16];
    a1.h[0] = *(const v8bf*)&s_p[ln][32 + hi * 8];
    a1.h[1] = *(const v8bf*)&s_p[ln][32 + hi * 8 + 16];
#pragma unroll
    for (int t = 0; t < 16; ++t) {
      int col = wv * 256 + t * 16 + ln;
      const __bf16* vrow = vT + ((size_t)bidx * DD + col) * SS + kb0 + hi * 8;
      V16 b0, b1;
      b0.h[0] = *(const v8bf*)(vrow);
      b0.h[1] = *(const v8bf*)(vrow + 16);
      b1.h[0] = *(const v8bf*)(vrow + 32);
      b1.h[1] = *(const v8bf*)(vrow + 48);
      acc[t] = wmma_bf16(a0.v, b0.v, acc[t]);
      acc[t] = wmma_bf16(a1.v, b1.v, acc[t]);
    }
    __syncthreads();  // protect s_p/s_max/s_sum reuse next iteration
  }

  // out = acc / l
#pragma unroll
  for (int e = 0; e < 8; ++e) {
    float inv = 1.0f / l_run[e];
    int row = qbase + e + hi * 8;
#pragma unroll
    for (int t = 0; t < 16; ++t) {
      int col = wv * 256 + t * 16 + ln;
      att[((size_t)bidx * SS + row) * DD + col] = acc[t][e] * inv;
    }
  }
}

// ---------------------------------------------------------------------------
// out_nxt = LN(x + att); writes f32 (residual path) and bf16 (next GEMM's A)
// ---------------------------------------------------------------------------
__global__ __launch_bounds__(128) void ln_res_kernel(
    const float* __restrict__ x, const float* __restrict__ att,
    const float* __restrict__ g, const float* __restrict__ bta,
    float* __restrict__ of, __bf16* __restrict__ ob) {
  const int row = blockIdx.x;
  const int tid = threadIdx.x;
  const float* xr = x + (size_t)row * DD;
  const float* ar = att + (size_t)row * DD;
  float v[4];
  float s = 0.0f, q = 0.0f;
#pragma unroll
  for (int i = 0; i < 4; ++i) {
    int c = tid + i * 128;
    v[i] = xr[c] + ar[c];
    s += v[i]; q += v[i] * v[i];
  }
#pragma unroll
  for (int m = 1; m < 32; m <<= 1) {
    s += __shfl_xor(s, m, 32);
    q += __shfl_xor(q, m, 32);
  }
  __shared__ float ss[4], sq[4];
  if ((tid & 31) == 0) { ss[tid >> 5] = s; sq[tid >> 5] = q; }
  __syncthreads();
  s = ss[0] + ss[1] + ss[2] + ss[3];
  q = sq[0] + sq[1] + sq[2] + sq[3];
  float mean = s / 512.0f;
  float var = q / 512.0f - mean * mean;
  float rstd = rsqrtf(var + 1e-5f);
  float* ofr = of + (size_t)row * DD;
  __bf16* obr = ob + (size_t)row * DD;
#pragma unroll
  for (int i = 0; i < 4; ++i) {
    int c = tid + i * 128;
    float o = (v[i] - mean) * rstd * g[c] + bta[c];
    ofr[c] = o;
    obr[c] = (__bf16)o;
  }
}

// ---------------------------------------------------------------------------
extern "C" void kernel_launch(void* const* d_in, const int* in_sizes, int n_in,
                              void* d_out, int out_size, void* d_ws,
                              size_t ws_size, hipStream_t stream) {
  const float* x = (const float*)d_in[0];
  const float* Wq = (const float*)d_in[1];
  const float* bq = (const float*)d_in[2];
  const float* Wk = (const float*)d_in[3];
  const float* bk = (const float*)d_in[4];
  const float* Wv = (const float*)d_in[5];
  const float* bv = (const float*)d_in[6];
  const float* ln0g = (const float*)d_in[7];
  const float* ln0b = (const float*)d_in[8];
  const float* W1 = (const float*)d_in[9];
  const float* b1 = (const float*)d_in[10];
  const float* ln1g = (const float*)d_in[11];
  const float* ln1b = (const float*)d_in[12];
  const float* W2 = (const float*)d_in[13];
  const float* b2 = (const float*)d_in[14];
  const float* ln2g = (const float*)d_in[15];
  const float* ln2b = (const float*)d_in[16];
  const float* W3 = (const float*)d_in[17];
  const float* b3 = (const float*)d_in[18];
  float* out = (float*)d_out;

  char* ws = (char*)d_ws;
  size_t off = 0;
  auto alloc = [&](size_t bytes) -> char* {
    char* p = ws + off;
    off += (bytes + 255) & ~(size_t)255;
    return p;
  };
  const size_t MD = (size_t)MM * DD;
  __bf16* x_b   = (__bf16*)alloc(MD * 2);
  __bf16* q_b   = (__bf16*)alloc(MD * 2);
  __bf16* k_b   = (__bf16*)alloc(MD * 2);
  __bf16* vT_b  = (__bf16*)alloc(MD * 2);
  float*  att   = (float*)alloc(MD * 4);
  float*  onx   = (float*)alloc(MD * 4);
  __bf16* onx_b = (__bf16*)alloc(MD * 2);
  __bf16* h1_b  = (__bf16*)alloc(MD * 2);
  __bf16* h2_b  = (__bf16*)alloc(MD * 2);
  const size_t WN = (size_t)DD * DD;
  __bf16* Wqt = (__bf16*)alloc(WN * 2);
  __bf16* Wkt = (__bf16*)alloc(WN * 2);
  __bf16* Wvt = (__bf16*)alloc(WN * 2);
  __bf16* W1t = (__bf16*)alloc(WN * 2);
  __bf16* W2t = (__bf16*)alloc(WN * 2);
  __bf16* W3t = (__bf16*)alloc(WN * 2);

  // --- conversions ---
  f32_to_bf16_kernel<<<(unsigned)((MD + 255) / 256), 256, 0, stream>>>(
      x, x_b, (long long)MD);
  unsigned wg = (unsigned)((WN + 255) / 256);
  transpose_bf16_kernel<<<wg, 256, 0, stream>>>(Wq, Wqt, DD, DD);
  transpose_bf16_kernel<<<wg, 256, 0, stream>>>(Wk, Wkt, DD, DD);
  transpose_bf16_kernel<<<wg, 256, 0, stream>>>(Wv, Wvt, DD, DD);
  transpose_bf16_kernel<<<wg, 256, 0, stream>>>(W1, W1t, DD, DD);
  transpose_bf16_kernel<<<wg, 256, 0, stream>>>(W2, W2t, DD, DD);
  transpose_bf16_kernel<<<wg, 256, 0, stream>>>(W3, W3t, DD, DD);

  dim3 gb(MM / 16), tb(256);
  // q, k (bf16 row-major), v (bf16 transposed [B][D][S])
  gemm_ep<<<gb, tb, 0, stream>>>(x_b, Wqt, bq, nullptr, nullptr, nullptr,
                                 q_b, nullptr, nullptr, MM, DD, DD, 0);
  gemm_ep<<<gb, tb, 0, stream>>>(x_b, Wkt, bk, nullptr, nullptr, nullptr,
                                 k_b, nullptr, nullptr, MM, DD, DD, 0);
  gemm_ep<<<gb, tb, 0, stream>>>(x_b, Wvt, bv, nullptr, nullptr, nullptr,
                                 nullptr, vT_b, nullptr, MM, DD, DD, 1);

  attn_kernel<<<dim3(BB * (SS / 16)), dim3(64), 0, stream>>>(q_b, k_b, vT_b, att);

  ln_res_kernel<<<dim3(MM), dim3(128), 0, stream>>>(x, att, ln0g, ln0b, onx, onx_b);

  gemm_ep<<<gb, tb, 0, stream>>>(onx_b, W1t, b1, onx, ln1g, ln1b,
                                 h1_b, nullptr, nullptr, MM, DD, DD, 2);
  gemm_ep<<<gb, tb, 0, stream>>>(h1_b, W2t, b2, onx, ln2g, ln2b,
                                 h2_b, nullptr, nullptr, MM, DD, DD, 2);
  gemm_ep<<<gb, tb, 0, stream>>>(h2_b, W3t, b3, nullptr, nullptr, nullptr,
                                 nullptr, nullptr, out, MM, DD, DD, 3);
}
